// LSTMActor_70497593197471
// MI455X (gfx1250) — compile-verified, matrix operand order checked by reading, not run on hardware
//
#include <hip/hip_runtime.h>
#include <hip/hip_bf16.h>
#include <stdint.h>

typedef __bf16 bf16;
typedef __bf16 v16bf __attribute__((ext_vector_type(16)));
typedef float  v8f   __attribute__((ext_vector_type(8)));
typedef int    v4i_  __attribute__((ext_vector_type(4)));
typedef unsigned int u32x4 __attribute__((ext_vector_type(4)));
typedef int    i32x8 __attribute__((ext_vector_type(8)));
typedef int    i32x4 __attribute__((ext_vector_type(4)));

#define TILE_M 128
#define TILE_N 128
#define TILE_K 64

// ---- CDNA5 async global->LDS + TDM paths (guarded: compile never fails) ---
#if defined(__has_builtin)
#  if __has_builtin(__builtin_amdgcn_global_load_async_to_lds_b128)
#    define HAVE_ASYNC_LDS 1
#  endif
#  if __has_builtin(__builtin_amdgcn_tensor_load_to_lds)
#    define HAVE_TDM 1
#  endif
#endif
#ifndef HAVE_ASYNC_LDS
#  define HAVE_ASYNC_LDS 0
#endif
#ifndef HAVE_TDM
#  define HAVE_TDM 0
#endif

__device__ __forceinline__ void async_wait0() {
#if HAVE_ASYNC_LDS
#  if __has_builtin(__builtin_amdgcn_s_wait_asynccnt)
    __builtin_amdgcn_s_wait_asynccnt(0);
#  else
    asm volatile("s_wait_asynccnt 0x0" ::: "memory");
#  endif
#endif
}

__device__ __forceinline__ void tensor_wait0() {
#if HAVE_TDM
#  if __has_builtin(__builtin_amdgcn_s_wait_tensorcnt)
    __builtin_amdgcn_s_wait_tensorcnt(0);
#  else
    asm volatile("s_wait_tensorcnt 0x0" ::: "memory");
#  endif
#endif
}

union Frag {
    uint4 u[2];
    v16bf v;
};

// ---------------------------------------------------------------------------
// fp32 -> bf16 elementwise convert (grid-stride)
// ---------------------------------------------------------------------------
__global__ void lstmactor_cvt_bf16(const float* __restrict__ in,
                                   bf16* __restrict__ out, size_t n) {
    size_t i = (size_t)blockIdx.x * blockDim.x + threadIdx.x;
    size_t stride = (size_t)gridDim.x * blockDim.x;
    for (; i < n; i += stride) out[i] = (bf16)in[i];
}

// bc = b_ih + b_hh
__global__ void lstmactor_bias_sum(const float* __restrict__ a,
                                   const float* __restrict__ b,
                                   float* __restrict__ o, int n) {
    int i = blockIdx.x * blockDim.x + threadIdx.x;
    if (i < n) o[i] = a[i] + b[i];
}

// ---------------------------------------------------------------------------
// bf16 WMMA GEMM: C[M,N] (fp32 acc) = A[M,K] @ B[K,N] (+bias)(+add)(relu)
// 256 threads = 8 waves; block tile 128x128, K tile 64, double-buffered LDS.
// Steady-state A tile staged by the Tensor Data Mover (one D# from wave 0,
// TENSORcnt); prologue A tile via async global->LDS (ASYNCcnt). B staged
// transposed via registers (global loads issued before the WMMA block).
// ---------------------------------------------------------------------------
__global__ __launch_bounds__(256) void lstmactor_gemm_bf16(
    const bf16* __restrict__ Ag, const bf16* __restrict__ Bg,
    const float* __restrict__ bias,   // [N] or nullptr
    const float* __restrict__ addm,   // [M,N] or nullptr (residual add)
    float* __restrict__ Cf,           // fp32 out or nullptr
    bf16* __restrict__ Cbf,           // bf16 out or nullptr
    int relu, int M, int N, int K) {
    __shared__ bf16 As[2][TILE_M * TILE_K];   // [m][k] row-major, 16 KB each
    __shared__ bf16 Bt[2][TILE_N * TILE_K];   // [n][k] transposed, 16 KB each

    const int tid    = threadIdx.x;
    const int lane   = tid & 31;
    const int wid    = tid >> 5;
    const int wm     = wid & 3;            // 4 waves across M (32 rows each)
    const int wn     = wid >> 2;           // 2 waves across N (64 cols each)
    const int l16    = lane & 15;
    const int rowAdd = (lane >> 4) << 3;   // C/D layout: upper lanes -> M+8
    const int kHalf  = (lane >> 4) << 3;   // A/B layout: upper lanes -> K+8

    const int blockN = blockIdx.x * TILE_N;
    const int blockM = blockIdx.y * TILE_M;

    const v8f vzero = {};
    v8f acc[2][4];
#pragma unroll
    for (int mt = 0; mt < 2; ++mt)
#pragma unroll
        for (int nt = 0; nt < 4; ++nt) acc[mt][nt] = vzero;

    // global->LDS staging coordinates
    const int arow = tid >> 1;                 // 0..127
    const int acol = (tid & 1) << 5;           // 0 or 32
    const bf16* aSrc = Ag + (size_t)(blockM + arow) * K + acol;

    const int brow = tid >> 2;                 // 0..63 (k within tile)
    const int bcol = (tid & 3) << 5;           // 0,32,64,96 (n within tile)
    const bf16* bSrc = Bg + (size_t)brow * N + blockN + bcol;

    // ---- tile staging helpers ----
    auto stageA = [&](int buf, int kk) {       // per-lane path (prologue)
        bf16* dst = &As[buf][arow * TILE_K + acol];
        const bf16* src = aSrc + kk;
#if HAVE_ASYNC_LDS
#pragma unroll
        for (int i = 0; i < 4; ++i)
            __builtin_amdgcn_global_load_async_to_lds_b128(
                (__attribute__((address_space(1))) v4i_*)(uintptr_t)(src + 8 * i),
                (__attribute__((address_space(3))) v4i_*)(dst + 8 * i), 0, 0);
#else
        const uint4* g = (const uint4*)src;
        uint4 t0 = g[0], t1 = g[1], t2 = g[2], t3 = g[3];
        *(uint4*)(dst + 0)  = t0;
        *(uint4*)(dst + 8)  = t1;
        *(uint4*)(dst + 16) = t2;
        *(uint4*)(dst + 24) = t3;
#endif
    };
#if HAVE_TDM
    // Tensor Data Mover: one 2D-tile descriptor covers the whole 128x64 A
    // tile (row-major, row stride K elements) -> contiguous LDS. Wave 0 only.
    auto stageA_tdm = [&](int buf, int kk) {
        if (wid == 0) {
            const uint64_t ga =
                (uint64_t)(uintptr_t)(Ag + (size_t)blockM * K + kk);
            const unsigned lds = (unsigned)(uintptr_t)(&As[buf][0]);
            u32x4 g0;
            g0[0] = 1u;                         // count=1 (valid user D#)
            g0[1] = lds;                        // lds_addr (bytes)
            g0[2] = (unsigned)ga;               // global_addr[31:0]
            g0[3] = (unsigned)((ga >> 32) & 0x01FFFFFFu) | (2u << 30); // [56:32], type=2
            i32x8 g1;
            g1[0] = (int)(1u << 16);            // data_size=1 -> 2 bytes
            g1[1] = (int)(((unsigned)K & 0xFFFFu) << 16);  // tensor_dim0 lo
            g1[2] = (int)((((unsigned)K >> 16) & 0xFFFFu) |
                          (((unsigned)M & 0xFFFFu) << 16)); // td0 hi | td1 lo
            g1[3] = (int)((((unsigned)M >> 16) & 0xFFFFu) |
                          ((unsigned)TILE_K << 16));        // td1 hi | tile_dim0
            g1[4] = (int)TILE_M;                // tile_dim1 | tile_dim2=0
            g1[5] = (int)K;                     // tensor_dim0_stride[31:0]
            g1[6] = 0;                          // stride0 hi | stride1 lo
            g1[7] = 0;                          // stride1 hi
            const i32x4 z4 = {};
            const i32x8 z8 = {};
            __builtin_amdgcn_tensor_load_to_lds(g0, g1, z4, z4, z8, 0);
        }
    };
#endif
    auto loadB = [&](uint4* r, int kk) {
        const uint4* g = (const uint4*)(bSrc + (size_t)kk * N);
        r[0] = g[0]; r[1] = g[1]; r[2] = g[2]; r[3] = g[3];
    };
    auto scatterB = [&](int buf, const uint4* r) {
        bf16 tmp[32];
        *(uint4*)&tmp[0]  = r[0];
        *(uint4*)&tmp[8]  = r[1];
        *(uint4*)&tmp[16] = r[2];
        *(uint4*)&tmp[24] = r[3];
#pragma unroll
        for (int e = 0; e < 32; ++e)
            Bt[buf][(bcol + e) * TILE_K + brow] = tmp[e];
    };
    auto compute = [&](int buf) {
#pragma unroll
        for (int ks = 0; ks < TILE_K; ks += 32) {
            Frag af[2], bfr[4];
#pragma unroll
            for (int mt = 0; mt < 2; ++mt) {
                const bf16* p =
                    &As[buf][(wm * 32 + mt * 16 + l16) * TILE_K + ks + kHalf];
                af[mt].u[0] = *(const uint4*)(p);
                af[mt].u[1] = *(const uint4*)(p + 16);  // K+16..23
            }
#pragma unroll
            for (int nt = 0; nt < 4; ++nt) {
                const bf16* p =
                    &Bt[buf][(wn * 64 + nt * 16 + l16) * TILE_K + ks + kHalf];
                bfr[nt].u[0] = *(const uint4*)(p);
                bfr[nt].u[1] = *(const uint4*)(p + 16);
            }
#pragma unroll
            for (int mt = 0; mt < 2; ++mt)
#pragma unroll
                for (int nt = 0; nt < 4; ++nt)
                    acc[mt][nt] = __builtin_amdgcn_wmma_f32_16x16x32_bf16(
                        false, af[mt].v, false, bfr[nt].v,
                        (short)0, acc[mt][nt], false, false);
        }
    };

    // ---- software pipeline: prologue (async path keeps ASYNCcnt coverage) --
    {
        stageA(0, 0);
        uint4 br[4];
        loadB(br, 0);
        scatterB(0, br);
        async_wait0();
        __syncthreads();
    }

    int buf = 0;
    for (int kk = 0; kk < K; kk += TILE_K) {
        const int nxt = buf ^ 1;
        const bool more = (kk + TILE_K) < K;
        uint4 br[4];
        if (more) {
#if HAVE_TDM
            stageA_tdm(nxt, kk + TILE_K);   // DMA engine fills As[nxt]
#else
            stageA(nxt, kk + TILE_K);
#endif
            loadB(br, kk + TILE_K);         // global->regs, overlaps WMMAs
            if (kk + 2 * TILE_K < K) {
                __builtin_prefetch(aSrc + kk + 2 * TILE_K, 0, 1);
                __builtin_prefetch(bSrc + (size_t)(kk + 2 * TILE_K) * N, 0, 1);
            }
        }
        compute(buf);                       // 16 x v_wmma from LDS buf
        if (more) {
            scatterB(nxt, br);
#if HAVE_TDM
            tensor_wait0();                 // wave0's TENSORcnt -> 0
#else
            async_wait0();
#endif
        }
        __syncthreads();
        buf = nxt;
    }

    // epilogue: lane l, vgpr r -> (M = base + r (+8 for lanes>=16), N = base + l16)
#pragma unroll
    for (int mt = 0; mt < 2; ++mt) {
#pragma unroll
        for (int nt = 0; nt < 4; ++nt) {
            const int mbase = blockM + wm * 32 + mt * 16 + rowAdd;
            const int ncol  = blockN + wn * 64 + nt * 16 + l16;
            float bv = bias ? bias[ncol] : 0.0f;
#pragma unroll
            for (int r = 0; r < 8; ++r) {
                const size_t idx = (size_t)(mbase + r) * N + ncol;
                float v = acc[mt][nt][r] + bv;
                if (addm) v += addm[idx];
                if (relu) v = fmaxf(v, 0.0f);
                if (Cf)  Cf[idx]  = v;
                if (Cbf) Cbf[idx] = (bf16)v;
            }
        }
    }
}

// ---------------------------------------------------------------------------
// LayerNorm + tanh, one block per row, bf16 output
// ---------------------------------------------------------------------------
__global__ __launch_bounds__(256) void lstmactor_ln_tanh(
    const float* __restrict__ x, const float* __restrict__ gamma,
    const float* __restrict__ beta, bf16* __restrict__ xbf, int Fd) {
    const int row = blockIdx.x;
    const float* xr = x + (size_t)row * Fd;
    float s = 0.0f, s2 = 0.0f;
    for (int j = threadIdx.x; j < Fd; j += 256) {
        float v = xr[j];
        s += v;
        s2 += v * v;
    }
    __shared__ float sh[256], sh2[256];
    sh[threadIdx.x] = s;
    sh2[threadIdx.x] = s2;
    __syncthreads();
    for (int st = 128; st > 0; st >>= 1) {
        if (threadIdx.x < st) {
            sh[threadIdx.x] += sh[threadIdx.x + st];
            sh2[threadIdx.x] += sh2[threadIdx.x + st];
        }
        __syncthreads();
    }
    const float mean = sh[0] / (float)Fd;
    const float var  = sh2[0] / (float)Fd - mean * mean;
    const float rstd = rsqrtf(var + 1e-5f);
    for (int j = threadIdx.x; j < Fd; j += 256) {
        float v = (xr[j] - mean) * rstd * gamma[j] + beta[j];
        xbf[(size_t)row * Fd + j] = (bf16)tanhf(v);
    }
}

// ---------------------------------------------------------------------------
// LSTM cell: gates [B,4H] (i|f|g|o) -> c (fp32 state), h -> bf16 into hs[t]
// ---------------------------------------------------------------------------
__device__ __forceinline__ float sigf(float x) {
    return 1.0f / (1.0f + __expf(-x));
}

__global__ void lstmactor_cell(const float* __restrict__ gates,
                               float* __restrict__ c,
                               bf16* __restrict__ hout, int first, int BH,
                               int Hd) {
    int idx = blockIdx.x * blockDim.x + threadIdx.x;
    if (idx >= BH) return;
    const int b = idx / Hd;
    const int j = idx - b * Hd;
    const float* g = gates + (size_t)b * 4 * Hd;
    const float gi = g[j];
    const float gf = g[Hd + j];
    const float gg = g[2 * Hd + j];
    const float go = g[3 * Hd + j];
    const float co = first ? 0.0f : c[idx];
    const float cn = sigf(gf) * co + sigf(gi) * tanhf(gg);
    const float h  = sigf(go) * tanhf(cn);
    c[idx] = cn;
    hout[idx] = (bf16)h;
}

// ---------------------------------------------------------------------------
// Head stage 2: out[b,t,a] = tanh( relu_z[t*B+b,:] @ W2[:,a] + b2[a] )
// ---------------------------------------------------------------------------
__global__ void lstmactor_head2(const bf16* __restrict__ z,
                                const float* __restrict__ W2,
                                const float* __restrict__ b2,
                                float* __restrict__ out, int T, int B, int A,
                                int Kd) {
    const int total = B * T * A;
    int idx = blockIdx.x * blockDim.x + threadIdx.x;
    if (idx >= total) return;
    const int a = idx % A;
    const int t = (idx / A) % T;
    const int b = idx / (A * T);
    const bf16* zr = z + ((size_t)t * B + b) * Kd;
    float acc = b2[a];
    for (int j = 0; j < Kd; ++j) acc += (float)zr[j] * W2[j * A + a];
    out[idx] = tanhf(acc);
}

// ---------------------------------------------------------------------------
// Host launcher
// ---------------------------------------------------------------------------
extern "C" void kernel_launch(void* const* d_in, const int* in_sizes, int n_in,
                              void* d_out, int out_size, void* d_ws,
                              size_t ws_size, hipStream_t stream) {
    (void)in_sizes; (void)n_in; (void)out_size; (void)ws_size;
    const int Bsz = 2048, R = 39200, Fd = 1024, Hd = 1024, Ad = 6, T = 16;
    const int H4 = 4 * Hd, Hh = Hd / 2;

    const float* obs     = (const float*)d_in[0];
    // d_in[1] = std (unused by reference)
    const float* W_trunk = (const float*)d_in[2];
    const float* b_trunk = (const float*)d_in[3];
    const float* gamma   = (const float*)d_in[4];
    const float* beta    = (const float*)d_in[5];
    const float* W_ih    = (const float*)d_in[6];
    const float* b_ih    = (const float*)d_in[7];
    const float* W_hh    = (const float*)d_in[8];
    const float* b_hh    = (const float*)d_in[9];
    const float* W1      = (const float*)d_in[10];
    const float* b1      = (const float*)d_in[11];
    const float* W2      = (const float*)d_in[12];
    const float* b2      = (const float*)d_in[13];
    float* out = (float*)d_out;

    char* base = (char*)d_ws;
    size_t off = 0;
    auto alloc = [&](size_t bytes) -> void* {
        void* p = base + off;
        off = (off + bytes + 255) & ~(size_t)255;
        return p;
    };
    bf16*  obs_bf = (bf16*)alloc((size_t)Bsz * R * 2);
    bf16*  Wt_bf  = (bf16*)alloc((size_t)R * Fd * 2);
    bf16*  Wih_bf = (bf16*)alloc((size_t)Fd * H4 * 2);
    bf16*  Whh_bf = (bf16*)alloc((size_t)Hd * H4 * 2);
    bf16*  W1_bf  = (bf16*)alloc((size_t)Hd * Hh * 2);
    float* bc     = (float*)alloc((size_t)H4 * 4);
    float* xf     = (float*)alloc((size_t)Bsz * Fd * 4);
    bf16*  xbf    = (bf16*)alloc((size_t)Bsz * Fd * 2);
    float* pre    = (float*)alloc((size_t)Bsz * H4 * 4);
    float* gates  = (float*)alloc((size_t)Bsz * H4 * 4);
    float* cst    = (float*)alloc((size_t)Bsz * Hd * 4);
    bf16*  hs_bf  = (bf16*)alloc((size_t)T * Bsz * Hd * 2);
    bf16*  z_bf   = (bf16*)alloc((size_t)T * Bsz * Hh * 2);

    // --- precision conversion of GEMM operands ---
    lstmactor_cvt_bf16<<<2048, 256, 0, stream>>>(obs, obs_bf, (size_t)Bsz * R);
    lstmactor_cvt_bf16<<<2048, 256, 0, stream>>>(W_trunk, Wt_bf, (size_t)R * Fd);
    lstmactor_cvt_bf16<<<1024, 256, 0, stream>>>(W_ih, Wih_bf, (size_t)Fd * H4);
    lstmactor_cvt_bf16<<<1024, 256, 0, stream>>>(W_hh, Whh_bf, (size_t)Hd * H4);
    lstmactor_cvt_bf16<<<256, 256, 0, stream>>>(W1, W1_bf, (size_t)Hd * Hh);
    lstmactor_bias_sum<<<(H4 + 255) / 256, 256, 0, stream>>>(b_ih, b_hh, bc, H4);

    // --- trunk: x = obs @ W_trunk + b_trunk ---
    {
        dim3 grid(Fd / TILE_N, Bsz / TILE_M);
        lstmactor_gemm_bf16<<<grid, 256, 0, stream>>>(
            obs_bf, Wt_bf, b_trunk, nullptr, xf, nullptr, 0, Bsz, Fd, R);
    }
    // --- LayerNorm + tanh -> bf16 ---
    lstmactor_ln_tanh<<<Bsz, 256, 0, stream>>>(xf, gamma, beta, xbf, Fd);

    // --- pre = x @ W_ih + (b_ih + b_hh) ---
    {
        dim3 grid(H4 / TILE_N, Bsz / TILE_M);
        lstmactor_gemm_bf16<<<grid, 256, 0, stream>>>(
            xbf, Wih_bf, bc, nullptr, pre, nullptr, 0, Bsz, H4, Fd);
    }

    // --- LSTM steps ---
    const int BH = Bsz * Hd;
    // t = 0: h0 == 0 -> gates == pre
    lstmactor_cell<<<(BH + 255) / 256, 256, 0, stream>>>(pre, cst, hs_bf, 1, BH,
                                                         Hd);
    for (int t = 1; t < T; ++t) {
        dim3 grid(H4 / TILE_N, Bsz / TILE_M);
        lstmactor_gemm_bf16<<<grid, 256, 0, stream>>>(
            hs_bf + (size_t)(t - 1) * BH, Whh_bf, nullptr, pre, gates, nullptr,
            0, Bsz, H4, Hd);
        lstmactor_cell<<<(BH + 255) / 256, 256, 0, stream>>>(
            gates, cst, hs_bf + (size_t)t * BH, 0, BH, Hd);
    }

    // --- head stage 1: z = relu(hs @ W1 + b1) -> bf16 ---
    {
        dim3 grid(Hh / TILE_N, (T * Bsz) / TILE_M);
        lstmactor_gemm_bf16<<<grid, 256, 0, stream>>>(
            hs_bf, W1_bf, b1, nullptr, nullptr, z_bf, 1, T * Bsz, Hh, Hd);
    }
    // --- head stage 2: mu = tanh(z @ W2 + b2), transpose to [B,T,A] ---
    {
        const int total = Bsz * T * Ad;
        lstmactor_head2<<<(total + 255) / 256, 256, 0, stream>>>(
            z_bf, W2, b2, out, T, Bsz, Ad, Hh);
    }
}
